// MyLSTM_75574244540894
// MI455X (gfx1250) — compile-verified, matrix operand order checked by reading, not run on hardware
//
#include <hip/hip_runtime.h>
#include <hip/hip_bf16.h>

// ---------------------------------------------------------------------------
// MI455X plan:
//  K1: emb LSTM forward (28 fused steps, Wih/Whh resident in LDS, WMMA f16,
//      double-buffered async x-tile staging via GLOBAL_LOAD_ASYNC_TO_LDS)
//  K1': emb LSTM backward = ONE step (only words[:, -1] is consumed)
//  K2: lin1 GEMM  [4096,278(288)] x [288,64] WMMA
//  K3: xproj for 2nd LSTM: [4096,50(64)] x [64,208] per direction, WMMA
//  K4: 2nd LSTM recurrence: 2 persistent blocks (dir), 128 steps, WMMA
//  K5: lin2 + log_softmax (tiny, VALU)
// ---------------------------------------------------------------------------

typedef __attribute__((ext_vector_type(16))) _Float16 v16h;
typedef __attribute__((ext_vector_type(8)))  _Float16 v8h;
typedef __attribute__((ext_vector_type(8)))  float    v8f;

#define DEVINL __device__ __forceinline__

// Load one 16x32 f16 WMMA operand fragment from LDS.
// Layout per CDNA5 ISA 7.12.2 (16-bit A 16x32): lanes 0-15 row m, halves
// {K[k0..k0+7], K[k0+16..k0+23]}; lanes 16-31 same rows, +8 K offset.
// B (32x16) mirrors this with lane = column (B consumed column-major).
DEVINL v16h lds_frag(const _Float16* base, int stride, int row0, int k0, int lane) {
  int r  = row0 + (lane & 15);
  int kb = k0 + ((lane >> 4) << 3);
  const _Float16* p = base + r * stride + kb;
  v8h lo = *(const v8h*)(p);
  v8h hi = *(const v8h*)(p + 16);
  return __builtin_shufflevector(lo, hi, 0,1,2,3,4,5,6,7,8,9,10,11,12,13,14,15);
}

DEVINL v8f wmma16(v16h a, v16h b, v8f c) {
  return __builtin_amdgcn_wmma_f32_16x16x32_f16(false, a, false, b,
                                                (short)0, c, false, false);
}

DEVINL float sigm(float x)  { return 1.f / (1.f + __expf(-x)); }
DEVINL float tanh_(float x) { return 2.f * sigm(2.f * x) - 1.f; }

// CDNA5 async global->LDS copy (ASYNCcnt path). vdst VGPR = LDS byte offset,
// vaddr = 64-bit global address (GV mode, SADDR=off).
DEVINL void async_ld_b32(unsigned int lds_off, const float* g) {
  asm volatile("global_load_async_to_lds_b32 %0, %1, off"
               :: "v"(lds_off), "v"((unsigned long long)(uintptr_t)g)
               : "memory");
}
DEVINL void wait_async0() {
  asm volatile("s_wait_asynccnt 0x0" ::: "memory");
}

// ---------------------------------------------------------------------------
// Kernel 1: emb LSTM, one direction, 32 sequences per block.
// x: [4096][206][113] f32. Gate GEMM per step:
//   G[32x208] = sX[32x128] * Wih^T  +  sH[32x64] * Whh^T  (+ bias)
// x tile for step t+1 is staged asynchronously while step t computes.
// ---------------------------------------------------------------------------
__global__ void __launch_bounds__(256)
lstm_emb_kernel(const float* __restrict__ x,
                const float* __restrict__ Wih,   // [200,113]
                const float* __restrict__ Whh,   // [200,50]
                const float* __restrict__ bih,
                const float* __restrict__ bhh,
                const float* __restrict__ h0,    // [4096,50] (dir slice)
                const float* __restrict__ c0,
                float* __restrict__ out_last,    // [4096,100]
                int out_off, int n_steps, int t0)
{
  __shared__ __align__(16) _Float16 sWih[208 * 136];
  __shared__ __align__(16) _Float16 sWhh[208 * 72];
  __shared__ __align__(16) _Float16 sX[32 * 136];
  __shared__ __align__(16) _Float16 sH[32 * 72];
  __shared__ __align__(16) float sXraw[32 * 128];   // async staging (f32)
  __shared__ float sC[32 * 52];
  __shared__ float sG[32 * 212];
  __shared__ float sBias[208];

  const int tid  = threadIdx.x;
  const int lane = tid & 31;
  const int wave = __builtin_amdgcn_readfirstlane(tid >> 5);
  const int seq0 = blockIdx.x * 32;
  const unsigned int rawBase = (unsigned int)(uintptr_t)(&sXraw[0]);

  // Prologue: kick off async stage of the first x tile; overlaps weight fill.
  for (int idx = tid; idx < 32 * 113; idx += 256) {
    int m = idx / 113, i = idx - m * 113;
    async_ld_b32(rawBase + (unsigned int)(m * 128 + i) * 4u,
                 &x[((seq0 + m) * 206 + t0) * 113 + i]);
  }

  for (int idx = tid; idx < 208 * 136; idx += 256) {
    int g = idx / 136, i = idx - g * 136;
    sWih[idx] = (g < 200 && i < 113) ? (_Float16)Wih[g * 113 + i] : (_Float16)0.f;
  }
  for (int idx = tid; idx < 208 * 72; idx += 256) {
    int g = idx / 72, k = idx - g * 72;
    sWhh[idx] = (g < 200 && k < 50) ? (_Float16)Whh[g * 50 + k] : (_Float16)0.f;
  }
  for (int idx = tid; idx < 208; idx += 256)
    sBias[idx] = (idx < 200) ? (bih[idx] + bhh[idx]) : 0.f;
  for (int idx = tid; idx < 32 * 72; idx += 256) {
    int m = idx / 72, k = idx - m * 72;
    sH[idx] = (k < 50) ? (_Float16)h0[(seq0 + m) * 50 + k] : (_Float16)0.f;
  }
  for (int idx = tid; idx < 32 * 52; idx += 256) {
    int m = idx / 52, j = idx - m * 52;
    sC[idx] = (j < 50) ? c0[(seq0 + m) * 50 + j] : 0.f;
  }

  const int mt  = wave & 1;   // 2 M-tiles
  const int nt0 = wave >> 1;  // 13 N-tiles striped by 4

  for (int step = 0; step < n_steps; ++step) {
    // Raw x tile for this step is in flight: wait + make visible to all waves.
    wait_async0();
    __syncthreads();

    // Convert f32 staging -> f16 WMMA operand tile (pad K 113->128).
    for (int idx = tid; idx < 32 * 128; idx += 256) {
      int m = idx >> 7, i = idx & 127;
      sX[m * 136 + i] = (i < 113) ? (_Float16)sXraw[m * 128 + i] : (_Float16)0.f;
    }
    __syncthreads();

    // Stage next step's x tile asynchronously behind the WMMA work.
    if (step + 1 < n_steps) {
      const int tn = t0 + step + 1;
      for (int idx = tid; idx < 32 * 113; idx += 256) {
        int m = idx / 113, i = idx - m * 113;
        async_ld_b32(rawBase + (unsigned int)(m * 128 + i) * 4u,
                     &x[((seq0 + m) * 206 + tn) * 113 + i]);
      }
    }

    v16h aX0 = lds_frag(sX, 136, mt * 16, 0,  lane);
    v16h aX1 = lds_frag(sX, 136, mt * 16, 32, lane);
    v16h aX2 = lds_frag(sX, 136, mt * 16, 64, lane);
    v16h aX3 = lds_frag(sX, 136, mt * 16, 96, lane);
    v16h aH0 = lds_frag(sH, 72,  mt * 16, 0,  lane);
    v16h aH1 = lds_frag(sH, 72,  mt * 16, 32, lane);
    for (int nt = nt0; nt < 13; nt += 4) {
      v8f acc = {};
      acc = wmma16(aX0, lds_frag(sWih, 136, nt * 16, 0,  lane), acc);
      acc = wmma16(aX1, lds_frag(sWih, 136, nt * 16, 32, lane), acc);
      acc = wmma16(aX2, lds_frag(sWih, 136, nt * 16, 64, lane), acc);
      acc = wmma16(aX3, lds_frag(sWih, 136, nt * 16, 96, lane), acc);
      acc = wmma16(aH0, lds_frag(sWhh, 72,  nt * 16, 0,  lane), acc);
      acc = wmma16(aH1, lds_frag(sWhh, 72,  nt * 16, 32, lane), acc);
      int col   = nt * 16 + (lane & 15);
      int mbase = mt * 16 + ((lane >> 4) << 3);
      #pragma unroll
      for (int r = 0; r < 8; ++r)
        sG[(mbase + r) * 212 + col] = acc[r] + sBias[col];
    }
    __syncthreads();

    const bool last = (step == n_steps - 1);
    for (int idx = tid; idx < 32 * 50; idx += 256) {
      int m = idx / 50, j = idx - m * 50;
      float ig = sigm(sG[m * 212 + j]);
      float fg = sigm(sG[m * 212 + 50 + j]);
      float gg = tanh_(sG[m * 212 + 100 + j]);
      float og = sigm(sG[m * 212 + 150 + j]);
      float c  = fg * sC[m * 52 + j] + ig * gg;
      float h  = og * tanh_(c);
      sC[m * 52 + j] = c;
      sH[m * 72 + j] = (_Float16)h;
      if (last) out_last[(seq0 + m) * 100 + out_off + j] = h;
    }
    __syncthreads();
  }
}

// ---------------------------------------------------------------------------
// Kernel 2: lin1.  bw[4096,50] = concat(bert[178], words[100]) @ W^T + b
// ---------------------------------------------------------------------------
__global__ void __launch_bounds__(256)
lin1_kernel(const float* __restrict__ x,
            const float* __restrict__ words,   // [4096,100]
            const float* __restrict__ W,       // [50,278]
            const float* __restrict__ b,       // [50]
            float* __restrict__ bw)            // [4096,50]
{
  __shared__ __align__(16) _Float16 sA[64 * 296];
  __shared__ __align__(16) _Float16 sB[64 * 296];
  __shared__ float sBias[64];
  const int tid = threadIdx.x, lane = tid & 31;
  const int wave = __builtin_amdgcn_readfirstlane(tid >> 5);
  const int seq0 = blockIdx.x * 64;

  for (int idx = tid; idx < 64 * 296; idx += 256) {
    int n = idx / 296, k = idx - n * 296;
    sB[idx] = (n < 50 && k < 278) ? (_Float16)W[n * 278 + k] : (_Float16)0.f;
  }
  for (int idx = tid; idx < 64 * 296; idx += 256) {
    int m = idx / 296, k = idx - m * 296;
    int seq = seq0 + m;
    float v = 0.f;
    if (k < 178)      v = x[((seq) * 206 + 28 + k) * 113 + 0];  // new_bert
    else if (k < 278) v = words[seq * 100 + (k - 178)];          // words_last
    sA[idx] = (_Float16)v;
  }
  for (int idx = tid; idx < 64; idx += 256)
    sBias[idx] = (idx < 50) ? b[idx] : 0.f;
  __syncthreads();

  const int mt = wave >> 1;                 // 4 M-tiles
  v16h aA[9];
  #pragma unroll
  for (int ks = 0; ks < 9; ++ks) aA[ks] = lds_frag(sA, 296, mt * 16, ks * 32, lane);
  for (int nt = (wave & 1) * 2; nt < (wave & 1) * 2 + 2; ++nt) {
    v8f acc = {};
    #pragma unroll
    for (int ks = 0; ks < 9; ++ks)
      acc = wmma16(aA[ks], lds_frag(sB, 296, nt * 16, ks * 32, lane), acc);
    int col   = nt * 16 + (lane & 15);
    int mbase = mt * 16 + ((lane >> 4) << 3);
    if (col < 50) {
      #pragma unroll
      for (int r = 0; r < 8; ++r)
        bw[(seq0 + mbase + r) * 50 + col] = acc[r] + sBias[col];
    }
  }
}

// ---------------------------------------------------------------------------
// Kernel 3: xproj for 2nd LSTM:  xp[dir][4096,200] = bw @ Wih^T + (bih+bhh)
// ---------------------------------------------------------------------------
__global__ void __launch_bounds__(256)
xproj_kernel(const float* __restrict__ bw,     // [4096,50]
             const float* __restrict__ Wih_f,  // [200,50]
             const float* __restrict__ Wih_b,
             const float* __restrict__ bih_f, const float* __restrict__ bhh_f,
             const float* __restrict__ bih_b, const float* __restrict__ bhh_b,
             float* __restrict__ xp)           // [2][4096][200]
{
  __shared__ __align__(16) _Float16 sA[64 * 72];
  __shared__ __align__(16) _Float16 sB[208 * 72];
  __shared__ float sBias[208];
  const int tid = threadIdx.x, lane = tid & 31;
  const int wave = __builtin_amdgcn_readfirstlane(tid >> 5);
  const int dir = blockIdx.y;
  const int row0 = blockIdx.x * 64;
  const float* W  = dir ? Wih_b : Wih_f;
  const float* b1 = dir ? bih_b : bih_f;
  const float* b2 = dir ? bhh_b : bhh_f;

  for (int idx = tid; idx < 208 * 72; idx += 256) {
    int n = idx / 72, k = idx - n * 72;
    sB[idx] = (n < 200 && k < 50) ? (_Float16)W[n * 50 + k] : (_Float16)0.f;
  }
  for (int idx = tid; idx < 64 * 72; idx += 256) {
    int m = idx / 72, k = idx - m * 72;
    sA[idx] = (k < 50) ? (_Float16)bw[(row0 + m) * 50 + k] : (_Float16)0.f;
  }
  for (int idx = tid; idx < 208; idx += 256)
    sBias[idx] = (idx < 200) ? (b1[idx] + b2[idx]) : 0.f;
  __syncthreads();

  const int mt = wave & 3, nt0 = wave >> 2;
  v16h a0 = lds_frag(sA, 72, mt * 16, 0,  lane);
  v16h a1 = lds_frag(sA, 72, mt * 16, 32, lane);
  float* out = xp + (size_t)dir * 4096 * 200;
  for (int nt = nt0; nt < 13; nt += 2) {
    v8f acc = {};
    acc = wmma16(a0, lds_frag(sB, 72, nt * 16, 0,  lane), acc);
    acc = wmma16(a1, lds_frag(sB, 72, nt * 16, 32, lane), acc);
    int col   = nt * 16 + (lane & 15);
    int mbase = mt * 16 + ((lane >> 4) << 3);
    if (col < 200) {
      #pragma unroll
      for (int r = 0; r < 8; ++r)
        out[(row0 + mbase + r) * 200 + col] = acc[r] + sBias[col];
    }
  }
}

// ---------------------------------------------------------------------------
// Kernel 4: 2nd LSTM recurrence, one persistent block per direction.
// ---------------------------------------------------------------------------
__global__ void __launch_bounds__(256)
lstm2_kernel(const float* __restrict__ xp,   // [2][4096][200]
             const float* __restrict__ Whh_f,
             const float* __restrict__ Whh_b,
             const float* __restrict__ h0,   // [2][32][50]
             const float* __restrict__ c0,
             float* __restrict__ out2)       // [32][128][100]
{
  __shared__ __align__(16) _Float16 sWhh[208 * 72];
  __shared__ __align__(16) _Float16 sH[32 * 72];
  __shared__ float sC[32 * 52];
  __shared__ float sG[32 * 212];
  const int tid = threadIdx.x, lane = tid & 31;
  const int wave = __builtin_amdgcn_readfirstlane(tid >> 5);
  const int dir = blockIdx.x;
  const float* Whh = dir ? Whh_b : Whh_f;
  const float* xpd = xp + (size_t)dir * 4096 * 200;

  for (int idx = tid; idx < 208 * 72; idx += 256) {
    int g = idx / 72, k = idx - g * 72;
    sWhh[idx] = (g < 200 && k < 50) ? (_Float16)Whh[g * 50 + k] : (_Float16)0.f;
  }
  for (int idx = tid; idx < 32 * 72; idx += 256) {
    int m = idx / 72, k = idx - m * 72;
    sH[idx] = (k < 50) ? (_Float16)h0[(dir * 32 + m) * 50 + k] : (_Float16)0.f;
  }
  for (int idx = tid; idx < 32 * 52; idx += 256) {
    int m = idx / 52, j = idx - m * 52;
    sC[idx] = (j < 50) ? c0[(dir * 32 + m) * 50 + j] : 0.f;
  }
  __syncthreads();

  const int mt = wave & 1, nt0 = wave >> 1;
  for (int t = 0; t < 128; ++t) {
    const int tt = dir ? (127 - t) : t;
    v16h a0 = lds_frag(sH, 72, mt * 16, 0,  lane);
    v16h a1 = lds_frag(sH, 72, mt * 16, 32, lane);
    for (int nt = nt0; nt < 13; nt += 4) {
      v8f acc = {};
      acc = wmma16(a0, lds_frag(sWhh, 72, nt * 16, 0,  lane), acc);
      acc = wmma16(a1, lds_frag(sWhh, 72, nt * 16, 32, lane), acc);
      int col   = nt * 16 + (lane & 15);
      int mbase = mt * 16 + ((lane >> 4) << 3);
      #pragma unroll
      for (int r = 0; r < 8; ++r) {
        int m = mbase + r;                               // m == batch index
        float xv = (col < 200) ? xpd[(m * 128 + tt) * 200 + col] : 0.f;
        sG[m * 212 + col] = acc[r] + xv;
      }
    }
    __syncthreads();
    for (int idx = tid; idx < 32 * 50; idx += 256) {
      int m = idx / 50, j = idx - m * 50;
      float ig = sigm(sG[m * 212 + j]);
      float fg = sigm(sG[m * 212 + 50 + j]);
      float gg = tanh_(sG[m * 212 + 100 + j]);
      float og = sigm(sG[m * 212 + 150 + j]);
      float c  = fg * sC[m * 52 + j] + ig * gg;
      float h  = og * tanh_(c);
      sC[m * 52 + j] = c;
      sH[m * 72 + j] = (_Float16)h;
      out2[(m * 128 + tt) * 100 + dir * 50 + j] = h;
    }
    __syncthreads();
  }
}

// ---------------------------------------------------------------------------
// Kernel 5: lin2 + log_softmax (tiny; one thread per row)
// ---------------------------------------------------------------------------
__global__ void __launch_bounds__(256)
head_kernel(const float* __restrict__ out2,   // [4096][100]
            const float* __restrict__ W,      // [9][100]
            const float* __restrict__ b,      // [9]
            float* __restrict__ out)          // [4096][9]
{
  __shared__ float sW[900];
  __shared__ float sb[9];
  const int tid = threadIdx.x;
  for (int idx = tid; idx < 900; idx += 256) sW[idx] = W[idx];
  if (tid < 9) sb[tid] = b[tid];
  __syncthreads();
  const int row = blockIdx.x * 256 + tid;
  float acc[9];
  #pragma unroll
  for (int k = 0; k < 9; ++k) acc[k] = sb[k];
  for (int i = 0; i < 100; ++i) {
    float v = out2[row * 100 + i];
    #pragma unroll
    for (int k = 0; k < 9; ++k) acc[k] += v * sW[k * 100 + i];
  }
  float mx = acc[0];
  #pragma unroll
  for (int k = 1; k < 9; ++k) mx = fmaxf(mx, acc[k]);
  float s = 0.f;
  #pragma unroll
  for (int k = 0; k < 9; ++k) s += __expf(acc[k] - mx);
  float lse = __logf(s);
  #pragma unroll
  for (int k = 0; k < 9; ++k) out[row * 9 + k] = acc[k] - mx - lse;
}

// ---------------------------------------------------------------------------
extern "C" void kernel_launch(void* const* d_in, const int* in_sizes, int n_in,
                              void* d_out, int out_size, void* d_ws, size_t ws_size,
                              hipStream_t stream) {
  const float* x      = (const float*)d_in[0];
  const float* eWih_f = (const float*)d_in[1];
  const float* eWhh_f = (const float*)d_in[2];
  const float* ebih_f = (const float*)d_in[3];
  const float* ebhh_f = (const float*)d_in[4];
  const float* eWih_b = (const float*)d_in[5];
  const float* eWhh_b = (const float*)d_in[6];
  const float* ebih_b = (const float*)d_in[7];
  const float* ebhh_b = (const float*)d_in[8];
  const float* lin1_W = (const float*)d_in[9];
  const float* lin1_b = (const float*)d_in[10];
  const float* Wih_f  = (const float*)d_in[11];
  const float* Whh_f  = (const float*)d_in[12];
  const float* bih_f  = (const float*)d_in[13];
  const float* bhh_f  = (const float*)d_in[14];
  const float* Wih_b  = (const float*)d_in[15];
  const float* Whh_b  = (const float*)d_in[16];
  const float* bih_b  = (const float*)d_in[17];
  const float* bhh_b  = (const float*)d_in[18];
  const float* lin2_W = (const float*)d_in[19];
  const float* lin2_b = (const float*)d_in[20];
  const float* h0_emb = (const float*)d_in[21];
  const float* c0_emb = (const float*)d_in[22];
  const float* h0     = (const float*)d_in[23];
  const float* c0     = (const float*)d_in[24];

  // Workspace layout (floats): words[409600] | bw[204800] | xp[1638400] | out2[409600]
  float* ws    = (float*)d_ws;
  float* words = ws;
  float* bw    = words + 4096 * 100;
  float* xp    = bw + 4096 * 50;
  float* out2  = xp + 2 * 4096 * 200;

  // emb LSTM forward: full 28 steps
  lstm_emb_kernel<<<128, 256, 0, stream>>>(x, eWih_f, eWhh_f, ebih_f, ebhh_f,
                                           h0_emb, c0_emb, words, 0, 28, 0);
  // emb LSTM backward at t=T-1 == one step on x[:,27,:]
  lstm_emb_kernel<<<128, 256, 0, stream>>>(x, eWih_b, eWhh_b, ebih_b, ebhh_b,
                                           h0_emb + 4096 * 50, c0_emb + 4096 * 50,
                                           words, 50, 1, 27);
  lin1_kernel<<<64, 256, 0, stream>>>(x, words, lin1_W, lin1_b, bw);
  xproj_kernel<<<dim3(64, 2), 256, 0, stream>>>(bw, Wih_f, Wih_b,
                                                bih_f, bhh_f, bih_b, bhh_b, xp);
  lstm2_kernel<<<2, 256, 0, stream>>>(xp, Whh_f, Whh_b, h0, c0, out2);
  head_kernel<<<16, 256, 0, stream>>>(out2, lin2_W, lin2_b, (float*)d_out);
}
